// Qwen2MoeDecoderLayer_76879914598703
// MI455X (gfx1250) — matrix operand order repaired and checked by reading, not compiled, not run on hardware
//
#include <hip/hip_runtime.h>
#include <hip/hip_bf16.h>
#include <math.h>

#define S_  2048
#define H_  2048
#define NH_ 16
#define D_  128
#define E_  8
#define IE_ 1408
#define IS_ 5632

typedef unsigned short u16;
typedef unsigned int   u32;
typedef __attribute__((ext_vector_type(16))) __bf16 v16bf;
typedef __attribute__((ext_vector_type(8)))  float  v8f;
typedef __attribute__((ext_vector_type(4)))  u32    v4u;
typedef __attribute__((ext_vector_type(4)))  float  v4f;

union Frag { v4u q[2]; v16bf v; };

__device__ __forceinline__ u16 f2bf(float f) {
    u32 u = __float_as_uint(f);
    u32 r = (u + 0x7FFFu + ((u >> 16) & 1u)) >> 16;
    return (u16)r;
}
__device__ __forceinline__ float bf2f(u16 h) {
    return __uint_as_float(((u32)h) << 16);
}

// ---------------------------------------------------------------------------
// RMSNorm: fp32 row -> bf16 row  (one block per token)
// ---------------------------------------------------------------------------
__global__ __launch_bounds__(256)
void rmsnorm_k(const float* __restrict__ x, const float* __restrict__ w,
               u16* __restrict__ out) {
    __shared__ float red[256];
    const int row = blockIdx.x, t = threadIdx.x;
    const float* xr = x + (size_t)row * H_;
    float ss = 0.f;
    for (int j = t; j < H_; j += 256) { float v = xr[j]; ss += v * v; }
    red[t] = ss; __syncthreads();
    for (int o = 128; o > 0; o >>= 1) {
        if (t < o) red[t] += red[t + o];
        __syncthreads();
    }
    const float inv = rsqrtf(red[0] * (1.0f / H_) + 1e-6f);
    for (int j = t; j < H_; j += 256)
        out[(size_t)row * H_ + j] = f2bf(xr[j] * inv * w[j]);
}

// ---------------------------------------------------------------------------
// WMMA GEMM: out(MxN) = bf16 A(MxK) x fp32 W(KxN), W converted to bf16 during
// LDS staging (transposed).  Block tile 128x128, 8 waves (4Mx2N), each wave
// 32x64 (8 WMMAs per K-step).  Double-buffered LDS: one barrier per K-step,
// next tile's global loads overlap the current WMMA chain.
//   EPI 0: out_bf16 = acc + bias[n]
//   EPI 1: out_f32  = acc + resid
//   EPI 2: out_f32  = acc
//   EPI 3: out_f32 += rowscale[m*rsStride] * acc     (MoE accumulate)
//   EPI 4: out_f32  = resid + addacc + rowscale[m] * acc  (final write)
// ---------------------------------------------------------------------------
template <int EPI>
__global__ __launch_bounds__(256)
void gemm_bf16_wmma(const u16* __restrict__ A, const float* __restrict__ W,
                    const float* __restrict__ bias, const float* __restrict__ resid,
                    const float* __restrict__ rowscale, int rsStride,
                    const float* __restrict__ addacc,
                    float* __restrict__ outF, u16* __restrict__ outB,
                    int M, int N, int K) {
    const int LDSR = 40;                 // padded row stride (elements)
    __shared__ u16 sA[2][128 * 40];      // A tile 128x32 bf16
    __shared__ u16 sB[2][128 * 40];      // W^T tile 128x32 bf16

    const int t    = threadIdx.x;
    const int wid  = t >> 5, lane = t & 31, lm = lane & 15, half = lane >> 4;
    const int wm   = (wid >> 1) * 32;    // wave row offset (0,32,64,96)
    const int wn   = (wid & 1) * 64;     // wave col offset (0,64)
    const int m0   = blockIdx.y * 128;
    const int n0   = blockIdx.x * 128;

    v8f acc[2][4];
#pragma unroll
    for (int ms = 0; ms < 2; ms++)
#pragma unroll
        for (int s = 0; s < 4; s++)
#pragma unroll
            for (int r = 0; r < 8; r++) acc[ms][s][r] = 0.f;

    const int ar = t >> 1, ac = (t & 1) * 16;  // A stage: 128 rows x 2 chunks
    const int bk = t >> 3, bn = (t & 7) * 16;  // W stage: 32 k-rows x 8 chunks

    // ---- tile staging helpers ----
    auto stageA = [&](int buf, int k0) {
        const u16* src = A + (size_t)(m0 + ar) * K + k0 + ac;
        *(v4u*)&sA[buf][ar * LDSR + ac]     = *(const v4u*)src;
        *(v4u*)&sA[buf][ar * LDSR + ac + 8] = *(const v4u*)(src + 8);
    };
    auto stageB = [&](int buf, int k0) {
        const v4f* src = (const v4f*)(W + (size_t)(k0 + bk) * N + n0 + bn);
        v4f f0 = src[0], f1 = src[1], f2 = src[2], f3 = src[3];
        float vals[16] = { f0.x, f0.y, f0.z, f0.w, f1.x, f1.y, f1.z, f1.w,
                           f2.x, f2.y, f2.z, f2.w, f3.x, f3.y, f3.z, f3.w };
#pragma unroll
        for (int i = 0; i < 16; i++)
            sB[buf][(bn + i) * LDSR + bk] = f2bf(vals[i]);
    };

    stageA(0, 0);
    stageB(0, 0);
    __syncthreads();

    int cur = 0;
    for (int k0 = 0; k0 < K; k0 += 32) {
        // stage next tile into the other buffer (overlaps WMMA below)
        if (k0 + 32 < K) {
            stageA(cur ^ 1, k0 + 32);
            stageB(cur ^ 1, k0 + 32);
            if (k0 + 64 < K) {  // cache hint two tiles ahead
                __builtin_prefetch(A + (size_t)(m0 + ar) * K + k0 + 64 + ac, 0, 1);
                __builtin_prefetch(W + (size_t)(k0 + 64 + bk) * N + n0 + bn, 0, 1);
            }
        }

        // A fragments for both 16-row sub-tiles (ISA bf16 A layout)
        Frag af[2];
#pragma unroll
        for (int ms = 0; ms < 2; ms++) {
            const u16* pa = &sA[cur][(wm + ms * 16 + lm) * LDSR + half * 8];
            af[ms].q[0] = *(const v4u*)pa;
            af[ms].q[1] = *(const v4u*)(pa + 16);
        }
#pragma unroll
        for (int s = 0; s < 4; s++) {
            Frag bf_;
            const u16* pb = &sB[cur][(wn + s * 16 + lm) * LDSR + half * 8];
            bf_.q[0] = *(const v4u*)pb;
            bf_.q[1] = *(const v4u*)(pb + 16);
#pragma unroll
            for (int ms = 0; ms < 2; ms++)
                acc[ms][s] = __builtin_amdgcn_wmma_f32_16x16x32_bf16(
                    false, af[ms].v, false, bf_.v, (short)0, acc[ms][s],
                    false, false);
        }
        __syncthreads();   // readers done with cur AND next-tile stores landed
        cur ^= 1;
    }

    // ---- epilogue (C layout: lane holds col lm, rows r + 8*half) ----
#pragma unroll
    for (int ms = 0; ms < 2; ms++) {
#pragma unroll
        for (int s = 0; s < 4; s++) {
#pragma unroll
            for (int r = 0; r < 8; r++) {
                const int m = m0 + wm + ms * 16 + r + 8 * half;
                const int n = n0 + wn + s * 16 + lm;
                const size_t o = (size_t)m * N + n;
                const float val = acc[ms][s][r];
                if (EPI == 0)      outB[o] = f2bf(val + bias[n]);
                else if (EPI == 1) outF[o] = val + resid[o];
                else if (EPI == 2) outF[o] = val;
                else if (EPI == 3) outF[o] += rowscale[(size_t)m * rsStride] * val;
                else               outF[o] = resid[o] + addacc[o] + rowscale[m] * val;
            }
        }
    }
}

// ---------------------------------------------------------------------------
// RoPE in place on bf16 q,k.  One thread per (s, head, pair i<64).
// ---------------------------------------------------------------------------
__global__ void rope_k(u16* __restrict__ q, u16* __restrict__ k) {
    const size_t idx = (size_t)blockIdx.x * 256 + threadIdx.x;
    if (idx >= (size_t)S_ * NH_ * 64) return;
    const int i = (int)(idx & 63);
    const size_t t2 = idx >> 6;
    const int head = (int)(t2 % NH_);
    const int s = (int)(t2 / NH_);
    const size_t base = (size_t)s * (NH_ * D_) + head * D_ + i;
    const float inv = __powf(1000000.0f, -(float)i * (1.0f / 64.0f));
    float sn, cs; __sincosf((float)s * inv, &sn, &cs);
    float a1 = bf2f(q[base]), a2 = bf2f(q[base + 64]);
    q[base]      = f2bf(a1 * cs - a2 * sn);
    q[base + 64] = f2bf(a2 * cs + a1 * sn);
    float b1 = bf2f(k[base]), b2 = bf2f(k[base + 64]);
    k[base]      = f2bf(b1 * cs - b2 * sn);
    k[base + 64] = f2bf(b2 * cs + b1 * sn);
}

// ---------------------------------------------------------------------------
// Flash attention (causal).  Grid (S/64, NH), block 128 = 4 waves x 16 q-rows.
// QK^T and P.V via v_wmma_f32_16x16x32_bf16; online softmax with width-16
// shuffles; V staged transposed in LDS; P re-laid out via per-wave LDS.
// ---------------------------------------------------------------------------
__global__ __launch_bounds__(128)
void flash_attn_k(const u16* __restrict__ qb, const u16* __restrict__ kb,
                  const u16* __restrict__ vb, u16* __restrict__ ob) {
    const int h  = blockIdx.y;
    const int q0 = blockIdx.x * 64;
    const int t = threadIdx.x, wid = t >> 5, lane = t & 31;
    const int lm = lane & 15, half = lane >> 4;
    const int ld = NH_ * D_;

    __shared__ u16 sVt[D_ * 40];      // V^T tile: 128 d-rows x 32 keys (pad 40)
    __shared__ u16 sP[4][16 * 40];    // per-wave P tile 16x32 (pad 40)

    const int qrow = q0 + wid * 16;

    // Q fragments for all 4 K-dim chunks (D=128)
    Frag qf[4];
#pragma unroll
    for (int c = 0; c < 4; c++) {
        const u16* p = qb + (size_t)(qrow + lm) * ld + h * D_ + c * 32;
        qf[c].q[0] = *(const v4u*)(p + half * 8);
        qf[c].q[1] = *(const v4u*)(p + 16 + half * 8);
    }

    v8f o[8];
    float mrow[8], lrow[8];
#pragma unroll
    for (int r = 0; r < 8; r++) {
        mrow[r] = -3.0e38f; lrow[r] = 0.f;
#pragma unroll
        for (int s = 0; s < 8; s++) o[s][r] = 0.f;
    }
    const float scale = 0.08838834764831845f;  // 1/sqrt(128)

    for (int kb0 = 0; kb0 < q0 + 64; kb0 += 32) {
        __syncthreads();
        {   // stage V^T: 32 keys x 128 dims -> sVt[d][key]
            const int key = t >> 2, d0 = (t & 3) * 32;
            const u16* p = vb + (size_t)(kb0 + key) * ld + h * D_ + d0;
            union { v4u q[4]; u16 s[32]; } vl;
            vl.q[0] = ((const v4u*)p)[0]; vl.q[1] = ((const v4u*)p)[1];
            vl.q[2] = ((const v4u*)p)[2]; vl.q[3] = ((const v4u*)p)[3];
#pragma unroll
            for (int i = 0; i < 32; i++) sVt[(d0 + i) * 40 + key] = vl.s[i];
        }
        __syncthreads();

        // scores: two 16-key tiles, K-dim = 128 (4 chained WMMAs each)
        v8f sc[2];
#pragma unroll
        for (int kt = 0; kt < 2; kt++) {
#pragma unroll
            for (int r = 0; r < 8; r++) sc[kt][r] = 0.f;
#pragma unroll
            for (int c = 0; c < 4; c++) {
                Frag kf;  // B^T row = key vector (contiguous)
                const u16* p = kb + (size_t)(kb0 + kt * 16 + lm) * ld + h * D_
                             + c * 32 + half * 8;
                kf.q[0] = *(const v4u*)p;
                kf.q[1] = *(const v4u*)(p + 16);
                sc[kt] = __builtin_amdgcn_wmma_f32_16x16x32_bf16(
                    false, qf[c].v, false, kf.v, (short)0, sc[kt], false, false);
            }
        }

        // online softmax, write P (bf16) to per-wave LDS in A-frag layout
#pragma unroll
        for (int r = 0; r < 8; r++) {
            const int qm = qrow + r + 8 * half;
            float v0 = sc[0][r] * scale; if (kb0 + lm > qm)      v0 = -3.0e38f;
            float v1 = sc[1][r] * scale; if (kb0 + 16 + lm > qm) v1 = -3.0e38f;
            float mx = fmaxf(v0, v1);
            for (int off = 1; off < 16; off <<= 1)
                mx = fmaxf(mx, __shfl_xor(mx, off, 16));
            const float mn = fmaxf(mrow[r], mx);
            const float alpha = __expf(mrow[r] - mn);
            mrow[r] = mn;
            const float p0 = __expf(v0 - mn), p1 = __expf(v1 - mn);
            float rs = p0 + p1;
            for (int off = 1; off < 16; off <<= 1)
                rs += __shfl_xor(rs, off, 16);
            lrow[r] = lrow[r] * alpha + rs;
#pragma unroll
            for (int s = 0; s < 8; s++) o[s][r] *= alpha;
            sP[wid][(r + 8 * half) * 40 + lm]      = f2bf(p0);
            sP[wid][(r + 8 * half) * 40 + 16 + lm] = f2bf(p1);
        }
        asm volatile("s_wait_dscnt 0" ::: "memory");  // P store -> P load RAW

        // O += P(16x32) x V(32x128)
        Frag pf;
        const u16* pp = &sP[wid][lm * 40 + half * 8];
        pf.q[0] = *(const v4u*)pp;
        pf.q[1] = *(const v4u*)(pp + 16);
#pragma unroll
        for (int s = 0; s < 8; s++) {
            Frag vf;
            const u16* pb = &sVt[(s * 16 + lm) * 40 + half * 8];
            vf.q[0] = *(const v4u*)pb;
            vf.q[1] = *(const v4u*)(pb + 16);
            o[s] = __builtin_amdgcn_wmma_f32_16x16x32_bf16(
                false, pf.v, false, vf.v, (short)0, o[s], false, false);
        }
    }

    // normalize and write bf16 output
#pragma unroll
    for (int r = 0; r < 8; r++) {
        const float inv = 1.f / lrow[r];
        const int m = qrow + r + 8 * half;
#pragma unroll
        for (int s = 0; s < 8; s++)
            ob[(size_t)m * ld + h * D_ + s * 16 + lm] = f2bf(o[s][r] * inv);
    }
}

// ---------------------------------------------------------------------------
// SiLU(g) * u -> bf16
// ---------------------------------------------------------------------------
__global__ void silu_mul_k(const float* __restrict__ g, const float* __restrict__ u,
                           u16* __restrict__ out, long n) {
    const long i = (long)blockIdx.x * 256 + threadIdx.x;
    if (i >= n) return;
    const float gv = g[i];
    out[i] = f2bf(gv / (1.f + __expf(-gv)) * u[i]);
}

// ---------------------------------------------------------------------------
// Router (+ shared-expert gate): logits, softmax, top-4 weights, sigmoid gate.
// One block per token.
// ---------------------------------------------------------------------------
__global__ __launch_bounds__(256)
void router_k(const u16* __restrict__ h2, const float* __restrict__ gw,
              const float* __restrict__ sgw, float* __restrict__ wout,
              float* __restrict__ sgv, float* __restrict__ logits_out) {
    __shared__ float red[9 * 256];
    const int row = blockIdx.x, t = threadIdx.x;
    float acc[9];
#pragma unroll
    for (int e = 0; e < 9; e++) acc[e] = 0.f;
    for (int j = t; j < H_; j += 256) {
        const float x = bf2f(h2[(size_t)row * H_ + j]);
#pragma unroll
        for (int e = 0; e < 8; e++) acc[e] += x * gw[(size_t)j * E_ + e];
        acc[8] += x * sgw[j];
    }
#pragma unroll
    for (int e = 0; e < 9; e++) red[e * 256 + t] = acc[e];
    __syncthreads();
    for (int o = 128; o > 0; o >>= 1) {
        if (t < o) {
#pragma unroll
            for (int e = 0; e < 9; e++) red[e * 256 + t] += red[e * 256 + t + o];
        }
        __syncthreads();
    }
    if (t == 0) {
        float lg[8], mx = -3.0e38f;
#pragma unroll
        for (int e = 0; e < 8; e++) { lg[e] = red[e * 256]; mx = fmaxf(mx, lg[e]); }
        float ex[8], sum = 0.f;
#pragma unroll
        for (int e = 0; e < 8; e++) { ex[e] = __expf(lg[e] - mx); sum += ex[e]; }
        float rw[8], wsel[8];
        bool used[8];
#pragma unroll
        for (int e = 0; e < 8; e++) { rw[e] = ex[e] / sum; wsel[e] = 0.f; used[e] = false; }
        for (int kk = 0; kk < 4; kk++) {
            int b = 0; float bv = -1.f;
            for (int e = 0; e < 8; e++)
                if (!used[e] && rw[e] > bv) { bv = rw[e]; b = e; }
            used[b] = true; wsel[b] = bv;
        }
        for (int e = 0; e < 8; e++) {
            wout[(size_t)row * E_ + e] = wsel[e];
            logits_out[(size_t)row * E_ + e] = lg[e];
        }
        sgv[row] = 1.f / (1.f + __expf(-red[8 * 256]));
    }
}

// ---------------------------------------------------------------------------
// Host launcher
// ---------------------------------------------------------------------------
extern "C" void kernel_launch(void* const* d_in, const int* in_sizes, int n_in,
                              void* d_out, int out_size, void* d_ws, size_t ws_size,
                              hipStream_t stream) {
    const float* X    = (const float*)d_in[0];
    const float* ln1  = (const float*)d_in[1];
    const float* ln2  = (const float*)d_in[2];
    const float* wq   = (const float*)d_in[3];
    const float* bq   = (const float*)d_in[4];
    const float* wk   = (const float*)d_in[5];
    const float* bk   = (const float*)d_in[6];
    const float* wv_w = (const float*)d_in[7];
    const float* bv   = (const float*)d_in[8];
    const float* wo   = (const float*)d_in[9];
    const float* gate = (const float*)d_in[10];
    const float* eg   = (const float*)d_in[11];
    const float* eu   = (const float*)d_in[12];
    const float* ed   = (const float*)d_in[13];
    const float* sgw  = (const float*)d_in[14];
    const float* suw  = (const float*)d_in[15];
    const float* sdw  = (const float*)d_in[16];
    const float* sgate= (const float*)d_in[17];

    // workspace layout (bytes, 256-aligned)
    char* w = (char*)d_ws;
    size_t off = 0;
    auto take = [&](size_t bytes) {
        void* p = w + off;
        off = (off + bytes + 255) & ~(size_t)255;
        return p;
    };
    u16*   hn   = (u16*)  take((size_t)S_ * H_ * 2);
    u16*   qb   = (u16*)  take((size_t)S_ * H_ * 2);
    u16*   kbuf = (u16*)  take((size_t)S_ * H_ * 2);
    u16*   vbuf = (u16*)  take((size_t)S_ * H_ * 2);
    u16*   attn = (u16*)  take((size_t)S_ * H_ * 2);
    float* xf   = (float*)take((size_t)S_ * H_ * 4);
    u16*   h2   = (u16*)  take((size_t)S_ * H_ * 2);
    float* moe  = (float*)take((size_t)S_ * H_ * 4);
    float* gf   = (float*)take((size_t)S_ * IS_ * 4);
    float* uf   = (float*)take((size_t)S_ * IS_ * 4);
    u16*   gup  = (u16*)  take((size_t)S_ * IS_ * 2);
    float* wtok = (float*)take((size_t)S_ * E_ * 4);
    float* sgv  = (float*)take((size_t)S_ * 4);

    float* out_x      = (float*)d_out;
    float* out_logits = (float*)d_out + (size_t)S_ * H_;

    const dim3 blk256(256), blk128(128);
    const dim3 gHH(H_ / 128, S_ / 128);   // N=2048 GEMMs
    const dim3 gIE(IE_ / 128, S_ / 128);  // N=1408
    const dim3 gIS(IS_ / 128, S_ / 128);  // N=5632

    // 1) RMSNorm1
    rmsnorm_k<<<S_, blk256, 0, stream>>>(X, ln1, hn);

    // 2) QKV projections (bias fused, bf16 out)
    gemm_bf16_wmma<0><<<gHH, blk256, 0, stream>>>(hn, wq, bq, nullptr, nullptr, 0,
        nullptr, nullptr, qb, S_, H_, H_);
    gemm_bf16_wmma<0><<<gHH, blk256, 0, stream>>>(hn, wk, bk, nullptr, nullptr, 0,
        nullptr, nullptr, kbuf, S_, H_, H_);
    gemm_bf16_wmma<0><<<gHH, blk256, 0, stream>>>(hn, wv_w, bv, nullptr, nullptr, 0,
        nullptr, nullptr, vbuf, S_, H_, H_);

    // 3) RoPE
    {
        const long n = (long)S_ * NH_ * 64;
        rope_k<<<(unsigned)((n + 255) / 256), blk256, 0, stream>>>(qb, kbuf);
    }

    // 4) flash attention
    flash_attn_k<<<dim3(S_ / 64, NH_), blk128, 0, stream>>>(qb, kbuf, vbuf, attn);

    // 5) output projection + residual -> x
    gemm_bf16_wmma<1><<<gHH, blk256, 0, stream>>>(attn, wo, nullptr, X, nullptr, 0,
        nullptr, xf, nullptr, S_, H_, H_);

    // 6) RMSNorm2
    rmsnorm_k<<<S_, blk256, 0, stream>>>(xf, ln2, h2);

    // 7) router + shared gate (writes router_logits into d_out tail)
    router_k<<<S_, blk256, 0, stream>>>(h2, gate, sgate, wtok, sgv, out_logits);

    // 8) MoE experts (dense, weighted accumulate)
    hipMemsetAsync(moe, 0, (size_t)S_ * H_ * 4, stream);
    for (int e = 0; e < E_; e++) {
        const float* egw = eg + (size_t)e * H_ * IE_;
        const float* euw = eu + (size_t)e * H_ * IE_;
        const float* edw = ed + (size_t)e * IE_ * H_;
        gemm_bf16_wmma<2><<<gIE, blk256, 0, stream>>>(h2, egw, nullptr, nullptr,
            nullptr, 0, nullptr, gf, nullptr, S_, IE_, H_);
        gemm_bf16_wmma<2><<<gIE, blk256, 0, stream>>>(h2, euw, nullptr, nullptr,
            nullptr, 0, nullptr, uf, nullptr, S_, IE_, H_);
        {
            const long n = (long)S_ * IE_;
            silu_mul_k<<<(unsigned)((n + 255) / 256), blk256, 0, stream>>>(gf, uf, gup, n);
        }
        gemm_bf16_wmma<3><<<gHH, blk256, 0, stream>>>(gup, edw, nullptr, nullptr,
            wtok + e, E_, nullptr, moe, nullptr, S_, H_, IE_);
    }

    // 9) shared expert + final combine -> d_out
    gemm_bf16_wmma<2><<<gIS, blk256, 0, stream>>>(h2, sgw, nullptr, nullptr,
        nullptr, 0, nullptr, gf, nullptr, S_, IS_, H_);
    gemm_bf16_wmma<2><<<gIS, blk256, 0, stream>>>(h2, suw, nullptr, nullptr,
        nullptr, 0, nullptr, uf, nullptr, S_, IS_, H_);
    {
        const long n = (long)S_ * IS_;
        silu_mul_k<<<(unsigned)((n + 255) / 256), blk256, 0, stream>>>(gf, uf, gup, n);
    }
    gemm_bf16_wmma<4><<<gHH, blk256, 0, stream>>>(gup, sdw, nullptr, xf,
        sgv, 1, moe, out_x, nullptr, S_, H_, IS_);
}
